// SplineConvBackbone_33131377721480
// MI455X (gfx1250) — matrix operand order repaired and checked by reading, not compiled, not run on hardware
//
#include <hip/hip_runtime.h>
#include <hip/hip_bf16.h>
#include <math.h>

// ---------------- problem constants ----------------
#define NNODES 20000
#define NEDGES 320000
#define CIN    128
#define HDIM   128
#define KKER   27        // 3^3 kernel matrices
#define STAPS  8         // 2^3 spline taps
#define AW     3584      // 27*128 + 128 (root) : folded GEMM K-width
#define KC_T   108       // K-chunks (of 32) sourced from T
#define KC_ALL 112       // + 4 chunks from root x
#define MTILES 1250      // 20000 / 16

typedef float  v8f   __attribute__((ext_vector_type(8)));
typedef __bf16 v16bf __attribute__((ext_vector_type(16)));

// pack two f32 -> packed bf16 (round-to-nearest via +0x8000)
__device__ __forceinline__ unsigned int pk_bf16(float lo, float hi) {
    unsigned int bl = __float_as_uint(lo);
    unsigned int bh = __float_as_uint(hi);
    return ((bh + 0x8000u) & 0xFFFF0000u) | ((bl + 0x8000u) >> 16);
}

// LDS swizzle: 16B chunk index rotated by row -> conflict-free b128 access
__device__ __forceinline__ int sw_idx(int row, int chunk) {   // in uint4 units
    return row * 16 + ((chunk + row) & 15);
}

// CDNA5 async global->LDS copy (ASYNCcnt); lds_off = wave-relative LDS byte address
__device__ __forceinline__ void async_ld_b128(unsigned lds_off, const void* gaddr) {
    asm volatile("global_load_async_to_lds_b128 %0, %1, off"
                 :: "v"(lds_off), "v"(gaddr) : "memory");
}
__device__ __forceinline__ void wait_asynccnt0() {
    asm volatile("s_wait_asynccnt 0x0" ::: "memory");
}
// CDNA5 packed-bf16 atomic add (no return) — L2 atomic units, 2 channels/op
__device__ __forceinline__ void atomic_pk_add_bf16(unsigned short* addr, unsigned int pk) {
    asm volatile("global_atomic_pk_add_bf16 %0, %1, off"
                 :: "v"(addr), "v"(pk) : "memory");
}

// ---------------- kernel 1: edge MLP + spline basis ----------------
__global__ void edge_prep_kernel(const float* __restrict__ ea,
                                 const int*   __restrict__ ei,
                                 const float* __restrict__ Wp1,
                                 const float* __restrict__ bp1,
                                 const float* __restrict__ Wp2,
                                 const float* __restrict__ bp2,
                                 float* __restrict__ bw,
                                 int*   __restrict__ seg) {
    int e = blockIdx.x * blockDim.x + threadIdx.x;
    if (e >= NEDGES) return;

    float a[16];
#pragma unroll
    for (int i = 0; i < 16; ++i) a[i] = ea[(size_t)e * 16 + i];

    float hid[6];
#pragma unroll
    for (int j = 0; j < 6; ++j) {
        float s = bp1[j];
#pragma unroll
        for (int i = 0; i < 16; ++i) s += a[i] * Wp1[i * 6 + j];
        hid[j] = fmaxf(s, 0.0f);
    }

    float frac[3]; int lo[3];
#pragma unroll
    for (int d = 0; d < 3; ++d) {
        float s = bp2[d];
#pragma unroll
        for (int j = 0; j < 6; ++j) s += hid[j] * Wp2[j * 3 + d];
        float u = 1.0f / (1.0f + expf(-s));     // sigmoid -> (0,1)
        float v = u * 2.0f;                      // * (KS-1)
        float l = floorf(v);
        l = fminf(fmaxf(l, 0.0f), 1.0f);         // clamp to [0, KS-2]
        frac[d] = v - l;
        lo[d]   = (int)l;
    }

    int dst = ei[NEDGES + e];
#pragma unroll
    for (int s = 0; s < STAPS; ++s) {
        float w = 1.0f; int idx = 0, stride = 1;
#pragma unroll
        for (int d = 0; d < 3; ++d) {
            int bit = (s >> d) & 1;
            w *= bit ? frac[d] : (1.0f - frac[d]);
            idx += (lo[d] + bit) * stride;
            stride *= 3;
        }
        bw[(size_t)e * STAPS + s]  = w;
        seg[(size_t)e * STAPS + s] = dst * KKER + idx;
    }
}

// ---------------- kernel 2: weights -> bf16, stacked [W ; Wr] per layer ----------------
__global__ void conv_weights_kernel(const float* __restrict__ W0, const float* __restrict__ Wr0,
                                    const float* __restrict__ W1, const float* __restrict__ Wr1,
                                    const float* __restrict__ W2, const float* __restrict__ Wr2,
                                    unsigned short* __restrict__ Wb) {
    int i = blockIdx.x * blockDim.x + threadIdx.x;   // over 3*AW*128 exactly
    int l = i / (AW * HDIM);
    int r = ((i % (AW * HDIM)) / 4096) * 32 + (i % 32);
    int c = (i / 32) % HDIM;
    const float* W  = (l == 0) ? W0  : (l == 1) ? W1  : W2;
    const float* Wr = (l == 0) ? Wr0 : (l == 1) ? Wr1 : Wr2;
    float f = (r < KKER * CIN) ? W[(size_t)r * HDIM + c]
                               : Wr[(size_t)(r - KKER * CIN) * HDIM + c];
    unsigned int b = __float_as_uint(f);
    unsigned int rb = b + 0x7FFFu + ((b >> 16) & 1u);   // round to nearest even
    Wb[i] = (unsigned short)(rb >> 16);
}

// ---------------- zero fill (16B granules) ----------------
__global__ void zero_f4_kernel(uint4* __restrict__ p, long n4) {
    long i = blockIdx.x * (long)blockDim.x + threadIdx.x;
    if (i < n4) p[i] = make_uint4(0u, 0u, 0u, 0u);
}

// ---------------- scatter: Tb[seg] += bf16(bw * x[src]) via pk-atomic ----------------
// 64 threads/edge, 2 channels per thread -> 4-byte aligned packed-bf16 atomics.
__global__ void scatter_pk_kernel(const float* __restrict__ xin,
                                  const int*   __restrict__ ei,
                                  const float* __restrict__ bw,
                                  const int*   __restrict__ seg,
                                  unsigned short* __restrict__ Tb) {
    int e  = blockIdx.x;          // one edge per block
    int c2 = threadIdx.x;         // 0..63 channel pair
    int srcn = ei[e];
    float xv0 = xin[(size_t)srcn * CIN + 2 * c2];
    float xv1 = xin[(size_t)srcn * CIN + 2 * c2 + 1];
    size_t base = (size_t)e * STAPS;
#pragma unroll
    for (int s = 0; s < STAPS; ++s) {
        float w  = bw[base + s];
        int   sg = seg[base + s];
        unsigned int pk = pk_bf16(w * xv0, w * xv1);
        atomic_pk_add_bf16(Tb + (size_t)sg * CIN + 2 * c2, pk);
    }
}

// ---------------- pack x -> bf16 (root-weight A columns) ----------------
__global__ void pack_x_kernel(const float* __restrict__ xin,
                              unsigned short* __restrict__ xb) {
    long t  = blockIdx.x * (long)blockDim.x + threadIdx.x;   // N*128/8 threads
    long j8 = t * 8;
    float4 f0 = *(const float4*)(xin + j8);
    float4 f1 = *(const float4*)(xin + j8 + 4);
    uint4 o;
    o.x = pk_bf16(f0.x, f0.y);
    o.y = pk_bf16(f0.z, f0.w);
    o.z = pk_bf16(f1.x, f1.y);
    o.w = pk_bf16(f1.z, f1.w);
    *(uint4*)(xb + j8) = o;
}

// ---------------- WMMA GEMM building blocks ----------------
__device__ __forceinline__ void prefetchB(const uint4* src, const unsigned* wrOff, unsigned nb) {
#pragma unroll
    for (int j = 0; j < 4; ++j)
        async_ld_b128(wrOff[j] + nb, src + 128 * j);
}

__device__ __forceinline__ void gemm_step(const unsigned short* sA,
                                          const char* bb, const unsigned* rdOff,
                                          v8f* acc) {
    union { v16bf v; uint4 q[2]; } A;
    A.q[0] = *(const uint4*)sA;
    A.q[1] = *(const uint4*)(sA + 16);
    union { v16bf v; uint4 q[2]; } B[4];
#pragma unroll
    for (int nt = 0; nt < 4; ++nt) {
        B[nt].q[0] = *(const uint4*)(bb + rdOff[2 * nt]);
        B[nt].q[1] = *(const uint4*)(bb + rdOff[2 * nt + 1]);
    }
#pragma unroll
    for (int nt = 0; nt < 4; ++nt)
        acc[nt] = __builtin_amdgcn_wmma_f32_16x16x32_bf16(
            false, A.v, false, B[nt].v, (short)0, acc[nt], false, false);
}

// ---------------- WMMA GEMM: out = bf16([Tb | xb]) @ Wb + b ----------
// Block = 128 threads = 4 waves = 2 M-tiles x 2 N-halves.
// B chunks (32x128 bf16 = 8KB) double-buffered in LDS via global_load_async_to_lds_b128.
__global__ __launch_bounds__(128) void spline_gemm_kernel(
        const unsigned short* __restrict__ Tb, // [N*3456] bf16
        const unsigned short* __restrict__ xb, // [N*128]  bf16
        const unsigned short* __restrict__ Wb, // [AW*128] bf16
        const float* __restrict__ bias,        // [128]
        float* __restrict__ out,               // [N*128]
        int doRelu) {
    __shared__ uint4 Bsh[2][512];              // 2 x 8KB

    const int tid   = threadIdx.x;
    const int wid   = tid >> 5;
    const int lane  = tid & 31;
    const int mtile = blockIdx.x * 2 + (wid >> 1);
    const int nhalf = wid & 1;                 // which 64 output columns

    const int mbase  = mtile * 16;
    const int rowM   = mbase + (lane & 15);
    const int hiHalf = lane >> 4;              // ISA 16-bit A layout (wave32)
    const int aOff   = hiHalf ? 8 : 0;

    v8f acc[4] = {};

    // lanes 0-15 read K 0-7 & 16-23; lanes 16-31 read K 8-15 & 24-31
    const unsigned short* ArowT = Tb + (size_t)rowM * (KKER * CIN) + aOff;
    const unsigned short* ArowX = xb + (size_t)rowM * CIN + aOff;

    const unsigned ldsBase = (unsigned)(unsigned long long)(const void*)&Bsh[0][0];
    const char*    bbase   = (const char*)&Bsh[0][0];
    const uint4*   Bg      = (const uint4*)Wb; // chunk kc = 512 uint4 starting at kc*512

    // per-thread async-copy slots: thread writes 4 x 16B per chunk
    const int wr_row = tid >> 4;
    const int wr_ch  = tid & 15;
    unsigned wrOff[4];
#pragma unroll
    for (int j = 0; j < 4; ++j)
        wrOff[j] = ldsBase + (unsigned)sw_idx(wr_row + 8 * j, wr_ch) * 16u;

    // per-lane B read offsets (8 x 16B fragments for this N-half)
    const int cBase = nhalf * 8;
    unsigned rdOff[8];
#pragma unroll
    for (int k = 0; k < 8; ++k) {
        int n = nhalf * 64 + (k >> 1) * 16 + (lane & 15);
        rdOff[k] = (unsigned)sw_idx(n >> 2, (n & 3) * 4 + (k & 1) * 2 + hiHalf) * 16u;
    }

    // preload chunk 0 into buffer 0
    prefetchB(Bg + tid, wrOff, 0u);
    wait_asynccnt0();
    __syncthreads();

    // hot loop: 108 T-chunks, prefetch of kc+1 always valid (KC_ALL > KC_T)
    for (int kc = 0; kc < KC_T; ++kc) {
        prefetchB(Bg + (size_t)(kc + 1) * 512 + tid, wrOff, (unsigned)((kc + 1) & 1) * 8192u);
        gemm_step(ArowT + kc * 32, bbase + (kc & 1) * 8192, rdOff, acc);
        wait_asynccnt0();      // my async writes to the next buffer have landed
        __syncthreads();       // all waves: done reading cur buf + done writing next buf
    }
    // tail: 4 root-x chunks (KC_T even -> buffer parity == k2 parity, compile-time)
#pragma unroll
    for (int k2 = 0; k2 < 4; ++k2) {
        const int kc = KC_T + k2;
        if (k2 < 3)
            prefetchB(Bg + (size_t)(kc + 1) * 512 + tid, wrOff, (unsigned)((k2 + 1) & 1) * 8192u);
        gemm_step(ArowX + k2 * 32, bbase + (k2 & 1) * 8192, rdOff, acc);
        if (k2 < 3) {
            wait_asynccnt0();
            __syncthreads();
        }
    }

    // C/D layout: VGPR r -> (M=r, N=lane) lanes 0-15 ; (M=r+8, N=lane-16) lanes 16-31
    const int colBase = nhalf * 64 + (lane & 15);
    const int rOff    = hiHalf * 8;
#pragma unroll
    for (int nt = 0; nt < 4; ++nt) {
        int col = colBase + nt * 16;
        float bv = bias[col];
#pragma unroll
        for (int r = 0; r < 8; ++r) {
            float v = acc[nt][r] + bv;
            if (doRelu) v = fmaxf(v, 0.0f);
            out[(size_t)(mbase + r + rOff) * HDIM + col] = v;
        }
    }
}

// ---------------- launcher ----------------
extern "C" void kernel_launch(void* const* d_in, const int* in_sizes, int n_in,
                              void* d_out, int out_size, void* d_ws, size_t ws_size,
                              hipStream_t stream) {
    const float* x   = (const float*)d_in[0];
    const int*   ei  = (const int*)  d_in[1];   // jnp int64 falls back to int32 w/o x64
    const float* ea  = (const float*)d_in[2];
    const float* Wp1 = (const float*)d_in[3];
    const float* bp1 = (const float*)d_in[4];
    const float* Wp2 = (const float*)d_in[5];
    const float* bp2 = (const float*)d_in[6];
    const float* Wl[3]  = { (const float*)d_in[7],  (const float*)d_in[10], (const float*)d_in[13] };
    const float* Wrl[3] = { (const float*)d_in[8],  (const float*)d_in[11], (const float*)d_in[14] };
    const float* bl[3]  = { (const float*)d_in[9],  (const float*)d_in[12], (const float*)d_in[15] };

    // workspace carve-up (all sizes multiples of 256B)
    char* ws = (char*)d_ws;
    float* bw = (float*)ws;                       ws += (size_t)NEDGES * STAPS * 4;   // 10.24 MB
    int*  seg = (int*)ws;                         ws += (size_t)NEDGES * STAPS * 4;   // 10.24 MB
    unsigned short* Wb = (unsigned short*)ws;     ws += (size_t)3 * AW * HDIM * 2;    // 2.75 MB
    float* h0 = (float*)ws;                       ws += (size_t)NNODES * HDIM * 4;    // 10.24 MB
    float* h1 = (float*)ws;                       ws += (size_t)NNODES * HDIM * 4;    // 10.24 MB
    unsigned short* xb = (unsigned short*)ws;     ws += (size_t)NNODES * HDIM * 2;    //  5.12 MB
    unsigned short* Tb = (unsigned short*)ws;     // N*27*128 bf16 = 138.2 MB (L2-resident)

    edge_prep_kernel<<<NEDGES / 256, 256, 0, stream>>>(ea, ei, Wp1, bp1, Wp2, bp2, bw, seg);
    conv_weights_kernel<<<(3 * AW * HDIM) / 256, 256, 0, stream>>>(
        Wl[0], Wrl[0], Wl[1], Wrl[1], Wl[2], Wrl[2], Wb);

    const long t_u4 = (long)NNODES * KKER * CIN * 2 / 16;   // 8,640,000 uint4 (bf16 T)
    float* hbuf[2] = { h0, h1 };
    const float* xin = x;
    for (int l = 0; l < 3; ++l) {
        zero_f4_kernel<<<(int)(t_u4 / 256), 256, 0, stream>>>((uint4*)Tb, t_u4);
        scatter_pk_kernel<<<NEDGES, CIN / 2, 0, stream>>>(xin, ei, bw, seg, Tb);
        pack_x_kernel<<<(NNODES * HDIM / 8) / 256, 256, 0, stream>>>(xin, xb);
        float* o = (l == 2) ? (float*)d_out : hbuf[l];
        spline_gemm_kernel<<<MTILES / 2, 128, 0, stream>>>(
            Tb, xb, Wb + (size_t)l * AW * HDIM, bl[l], o, (l < 2) ? 1 : 0);
        xin = o;
    }
}
